// CrossAttention2D_49478023250656
// MI455X (gfx1250) — matrix operand order repaired and checked by reading, not compiled, not run on hardware
//
#include <hip/hip_runtime.h>
#include <hip/hip_bf16.h>

typedef __attribute__((ext_vector_type(16))) _Float16 v16h;
typedef __attribute__((ext_vector_type(8)))  _Float16 v8h;
typedef __attribute__((ext_vector_type(8)))  float    v8f;
typedef __attribute__((ext_vector_type(4)))  int      v4i;

// address-space-qualified views for the async-LDS builtin
typedef __attribute__((address_space(1))) v4i v4i_as1;
typedef __attribute__((address_space(3))) v4i v4i_as3;

#define WMMA_F16(a, b, c) \
  __builtin_amdgcn_wmma_f32_16x16x32_f16(false, (a), false, (b), (short)0, (c), false, false)

#if defined(__AMDGCN__) && __has_builtin(__builtin_amdgcn_global_load_async_to_lds_b128) && \
    __has_builtin(__builtin_amdgcn_s_wait_asynccnt)
#define USE_ASYNC_LDS 1
#else
#define USE_ASYNC_LDS 0
#endif

__device__ __forceinline__ v16h cat16(v8h lo, v8h hi) {
  return __builtin_shufflevector(lo, hi, 0, 1, 2, 3, 4, 5, 6, 7,
                                         8, 9, 10, 11, 12, 13, 14, 15);
}

// 16-byte global -> LDS copy; async (no VGPR round trip) when available.
__device__ __forceinline__ void g2l_b128(_Float16* dst, const _Float16* src) {
#if USE_ASYNC_LDS
  __builtin_amdgcn_global_load_async_to_lds_b128(
      (v4i_as1*)src, (v4i_as3*)dst, 0, 0);
#else
  *(v8h*)dst = *(const v8h*)src;
#endif
}

__device__ __forceinline__ void g2l_fence() {
#if USE_ASYNC_LDS
  __builtin_amdgcn_s_wait_asynccnt(0);
#endif
}

constexpr int Bn  = 8;
constexpr int Cn  = 256;
constexpr int HWn = 4096;

// LDS row strides in halves; multiples of 8 (16B alignment for b128 loads),
// dword strides have gcd(.,64)==4 -> conflict-free-ish b128 access.
constexpr int LXS = 72;    // proj x tile rows: 64 ch + 8 pad
constexpr int LWS = 72;    // proj w tile rows: 64 ch + 8 pad
constexpr int LKC = 264;   // attn K tile rows: 256 ch + 8 pad
constexpr int LVS = 40;    // attn V^T tile rows: 32 keys + 8 pad
constexpr int LPS = 40;    // attn P tile rows: 32 keys + 8 pad

// ---------------------------------------------------------------------------
// Kernel 1: fused 1x1-conv projections (q/k/v selected by blockIdx.z).
//   out[n,o] = sum_c x[c,n] * w[o,c] + b[o]
// Q,K stored [B, HW, C] f16;  V stored [B, C, HW] f16 (transposed for PV).
// ---------------------------------------------------------------------------
__global__ __launch_bounds__(256) void proj_kernel(
    const float* __restrict__ xq, const float* __restrict__ xk, const float* __restrict__ xv,
    const float* __restrict__ wq, const float* __restrict__ wk, const float* __restrict__ wv,
    const float* __restrict__ bq, const float* __restrict__ bk, const float* __restrict__ bv,
    _Float16* __restrict__ Qo, _Float16* __restrict__ Ko, _Float16* __restrict__ Vo)
{
  const int which = blockIdx.z;
  const float* __restrict__ x    = (which == 0) ? xq : (which == 1) ? xk : xv;
  const float* __restrict__ w    = (which == 0) ? wq : (which == 1) ? wk : wv;
  const float* __restrict__ bias = (which == 0) ? bq : (which == 1) ? bk : bv;

  const int b    = blockIdx.y;
  const int n0   = blockIdx.x * 128;
  const int tid  = threadIdx.x;
  const int lane = tid & 31;
  const int wv_  = tid >> 5;     // wave id 0..7 -> 16 output rows each
  const int lg   = lane >> 4;    // lane group
  const int ln   = lane & 15;

  __shared__ alignas(16) _Float16 lds_x[128 * LXS];  // [n][c] (A operand, row-major)
  __shared__ alignas(16) _Float16 lds_w[256 * LWS];  // [o][c] (B^T layout: col-major of B)

  v8f acc[16] = {};

  const float* xb = x + (size_t)b * Cn * HWn;

  auto loadW = [&](int t, int cb) -> v16h {
    const int o = t * 16 + ln;
    return cat16(*(const v8h*)&lds_w[o * LWS + cb + 8 * lg],
                 *(const v8h*)&lds_w[o * LWS + cb + 16 + 8 * lg]);
  };

  for (int c0 = 0; c0 < Cn; c0 += 64) {
    __syncthreads();
    // stage x^T tile [128 n x 64 c]; global reads contiguous along n
    for (int i = tid; i < 128 * 64; i += 256) {
      int cc = i >> 7;
      int nn = i & 127;
      lds_x[nn * LXS + cc] = (_Float16)xb[(size_t)(c0 + cc) * HWn + (n0 + nn)];
    }
    // stage w tile [256 o x 64 c]; both sides contiguous along c
    for (int i = tid; i < 256 * 64; i += 256) {
      int oo = i >> 6;
      int cc = i & 63;
      lds_w[oo * LWS + cc] = (_Float16)w[oo * Cn + (c0 + cc)];
    }
    __syncthreads();

    #pragma unroll
    for (int sc = 0; sc < 2; ++sc) {     // two K=32 sub-chunks
      const int cb = sc * 32;
      const int m  = wv_ * 16 + ln;
      // A fragment: two 16B runs  (K = cb+8g..+7 , cb+16+8g..+7)
      v16h a = cat16(*(const v8h*)&lds_x[m * LXS + cb + 8 * lg],
                     *(const v8h*)&lds_x[m * LXS + cb + 16 + 8 * lg]);
      // software-pipelined B-fragment stream: load t+1 before WMMA t
      v16h bm = loadW(0, cb);
      #pragma unroll
      for (int t = 0; t < 16; ++t) {
        v16h bnext = (t < 15) ? loadW(t + 1, cb) : bm;
        acc[t] = WMMA_F16(a, bm, acc[t]);
        bm = bnext;
      }
    }
  }

  // epilogue: bias add, f16 convert, layout-specific store
  #pragma unroll
  for (int t = 0; t < 16; ++t) {
    const int o    = t * 16 + ln;
    const float bo = bias[o];
    #pragma unroll
    for (int r = 0; r < 8; ++r) {
      const int n     = n0 + wv_ * 16 + r + 8 * lg;
      const float val = acc[t][r] + bo;
      if (which == 2) {
        Vo[(size_t)b * Cn * HWn + (size_t)o * HWn + n] = (_Float16)val;   // V^T: [C, HW]
      } else {
        _Float16* dst = (which == 0) ? Qo : Ko;
        dst[(size_t)b * HWn * Cn + (size_t)n * Cn + o] = (_Float16)val;   // [HW, C]
      }
    }
  }
}

// ---------------------------------------------------------------------------
// Kernel 2: flash attention. Each wave owns 16 query rows and all 256 output
// channels; loops over key blocks of 32 with online softmax.
// ---------------------------------------------------------------------------
__global__ __launch_bounds__(256) void attn_kernel(
    const _Float16* __restrict__ Q, const _Float16* __restrict__ K,
    const _Float16* __restrict__ V, float* __restrict__ out)
{
  const int b    = blockIdx.y;
  const int q0   = blockIdx.x * 128;
  const int tid  = threadIdx.x;
  const int lane = tid & 31;
  const int wv_  = tid >> 5;
  const int lg   = lane >> 4;
  const int ln   = lane & 15;

  __shared__ alignas(16) _Float16 lds_k[32 * LKC];       // K tile  [key][ch]
  __shared__ alignas(16) _Float16 lds_vt[256 * LVS];     // V^T tile [ch][key]
  __shared__ alignas(16) _Float16 lds_p[8][16 * LPS];    // per-wave P staging [m][key]

  // resident Q slab: 16 rows x 256 channels as 8 A-fragments (2x b128 each)
  const _Float16* Qb = Q + (size_t)b * HWn * Cn + (size_t)(q0 + wv_ * 16) * Cn;
  v16h qa[8];
  #pragma unroll
  for (int cc = 0; cc < 8; ++cc) {
    const _Float16* qrow = Qb + (size_t)ln * Cn + cc * 32;
    qa[cc] = cat16(*(const v8h*)(qrow + 8 * lg),
                   *(const v8h*)(qrow + 16 + 8 * lg));
  }

  v8f oacc[16] = {};
  float mstate[8], lstate[8];
  #pragma unroll
  for (int r = 0; r < 8; ++r) { mstate[r] = -3.0e38f; lstate[r] = 0.f; }

  const _Float16* Kb = K + (size_t)b * HWn * Cn;   // [HW, C]
  const _Float16* Vb = V + (size_t)b * Cn * HWn;   // [C, HW]

  auto loadK = [&](int cc, int joff) -> v16h {
    const _Float16* kr = &lds_k[(joff + ln) * LKC + cc * 32 + 8 * lg];
    return cat16(*(const v8h*)kr, *(const v8h*)(kr + 16));
  };
  auto loadV = [&](int t) -> v16h {
    const _Float16* vr = &lds_vt[(t * 16 + ln) * LVS + 8 * lg];
    return cat16(*(const v8h*)vr, *(const v8h*)(vr + 16));
  };

  for (int k0 = 0; k0 < HWn; k0 += 32) {
    __syncthreads();   // previous iteration done reading lds_k / lds_vt
    // stage K tile [32 key x 256 ch]; contiguous 16B transfers both sides
    for (int i = tid; i < 32 * 32; i += 256) {
      int j  = i >> 5;
      int cg = (i & 31) * 8;
      g2l_b128(&lds_k[j * LKC + cg], &Kb[(size_t)(k0 + j) * Cn + cg]);
    }
    // stage V^T tile [256 ch x 32 key]; contiguous 16B transfers both sides
    for (int i = tid; i < 256 * 4; i += 256) {
      int c  = i >> 2;
      int kg = (i & 3) * 8;
      g2l_b128(&lds_vt[c * LVS + kg], &Vb[(size_t)c * HWn + k0 + kg]);
    }
    if (k0 + 32 < HWn) {   // prefetch next tiles into cache
      __builtin_prefetch(&Kb[(size_t)(k0 + 32 + (tid >> 3)) * Cn + (tid & 7) * 32], 0, 0);
      __builtin_prefetch(&Vb[(size_t)tid * HWn + k0 + 32], 0, 0);
    }
    g2l_fence();     // own async transfers done (ASYNCcnt==0)
    __syncthreads(); // all waves' transfers done

    // ---- S = Q K^T : two 16x16 tiles (keys k0..+15, k0+16..+31) ----
    v8f s0 = {}, s1 = {};
    {
      v16h b0 = loadK(0, 0), b1 = loadK(0, 16);
      #pragma unroll
      for (int cc = 0; cc < 8; ++cc) {
        v16h n0_ = b0, n1_ = b1;
        if (cc < 7) { n0_ = loadK(cc + 1, 0); n1_ = loadK(cc + 1, 16); }
        s0 = WMMA_F16(qa[cc], b0, s0);
        s1 = WMMA_F16(qa[cc], b1, s1);
        b0 = n0_; b1 = n1_;
      }
    }

    // ---- online softmax (row = r + 8*lg, cols across 16 lanes) ----
    #pragma unroll
    for (int r = 0; r < 8; ++r) {
      float tmax = fmaxf(s0[r], s1[r]);
      #pragma unroll
      for (int msk = 1; msk < 16; msk <<= 1)
        tmax = fmaxf(tmax, __shfl_xor(tmax, msk, 32));
      const float nm    = fmaxf(mstate[r], tmax);
      const float scale = __expf(mstate[r] - nm);
      const float p0 = __expf(s0[r] - nm);
      const float p1 = __expf(s1[r] - nm);
      float psum = p0 + p1;
      #pragma unroll
      for (int msk = 1; msk < 16; msk <<= 1)
        psum += __shfl_xor(psum, msk, 32);
      lstate[r] = lstate[r] * scale + psum;
      mstate[r] = nm;
      s0[r] = p0;
      s1[r] = p1;
      #pragma unroll
      for (int t = 0; t < 16; ++t) oacc[t][r] *= scale;
    }

    // ---- re-layout P (C/D layout -> A-fragment layout) through LDS ----
    #pragma unroll
    for (int r = 0; r < 8; ++r) {
      lds_p[wv_][(r + 8 * lg) * LPS + ln]      = (_Float16)s0[r];
      lds_p[wv_][(r + 8 * lg) * LPS + 16 + ln] = (_Float16)s1[r];
    }
    __syncthreads();
    v16h pa = cat16(*(const v8h*)&lds_p[wv_][ln * LPS + 8 * lg],
                    *(const v8h*)&lds_p[wv_][ln * LPS + 16 + 8 * lg]);

    // ---- O += P V  (software-pipelined B-fragment stream) ----
    {
      v16h bm = loadV(0);
      #pragma unroll
      for (int t = 0; t < 16; ++t) {
        v16h bnext = (t < 15) ? loadV(t + 1) : bm;
        oacc[t] = WMMA_F16(pa, bm, oacc[t]);
        bm = bnext;
      }
    }
  }

  // ---- finalize: divide by row sums, store transposed to [B, C, H, W] ----
  float* outb = out + (size_t)b * Cn * HWn;
  #pragma unroll
  for (int t = 0; t < 16; ++t) {
    const int ch = t * 16 + ln;
    #pragma unroll
    for (int r = 0; r < 8; ++r) {
      const int n = q0 + wv_ * 16 + r + 8 * lg;
      outb[(size_t)ch * HWn + n] = oacc[t][r] * (1.0f / lstate[r]);
    }
  }
}

// ---------------------------------------------------------------------------
extern "C" void kernel_launch(void* const* d_in, const int* in_sizes, int n_in,
                              void* d_out, int out_size, void* d_ws, size_t ws_size,
                              hipStream_t stream) {
  const float* query = (const float*)d_in[0];
  const float* key   = (const float*)d_in[1];
  const float* value = (const float*)d_in[2];
  const float* wq    = (const float*)d_in[3];
  const float* bq    = (const float*)d_in[4];
  const float* wk    = (const float*)d_in[5];
  const float* bk    = (const float*)d_in[6];
  const float* wv    = (const float*)d_in[7];
  const float* bv    = (const float*)d_in[8];

  const size_t per_tensor = (size_t)Bn * HWn * Cn;   // 8,388,608 halves
  _Float16* Q = (_Float16*)d_ws;
  _Float16* K = Q + per_tensor;
  _Float16* V = K + per_tensor;

  // Phase 1: q/k/v projections (f16 outputs in workspace)
  proj_kernel<<<dim3(HWn / 128, Bn, 3), 256, 0, stream>>>(
      query, key, value, wq, wk, wv, bq, bk, bv, Q, K, V);

  // Phase 2: flash attention, fp32 output in reference layout
  attn_kernel<<<dim3(HWn / 128, Bn), 256, 0, stream>>>(Q, K, V, (float*)d_out);
}
